// GraphAttention_55490977465189
// MI455X (gfx1250) — compile-verified, matrix operand order checked by reading, not run on hardware
//
#include <hip/hip_runtime.h>

// GAT layer (sigmoid attention, custom normalization), MI455X / gfx1250.
// N=1024 nodes, F=128 in-features, Fp=64 per head, H=2 heads.
//
// Pipeline:
//   1) gat_proj : HW1/HW2 = X @ W{1,2} via v_wmma_f32_16x16x32_bf16.
//                 Also emits HW2^T in bf16 [Fp][N] so the aggregation GEMM's
//                 B-fragments become contiguous 16B loads.
//   2) gat_att  : edge-masked pairwise logits (VALU, ~5% of pairs), sigmoid,
//                 row-normalize; writes final att f32 (output #2), att bf16
//                 (A-operand for the agg GEMM) and beta.
//   3) gat_agg  : out = relu(beta*HW1 + att @ HW2), bf16 WMMA, K=1024.
//                 B tile staged in LDS via async global->LDS copies
//                 (ASYNCcnt / s_wait_asynccnt), double-buffered.

#define NN 1024
#define FF 128
#define FP 64
#define HH 2
#define SLOPE 0.2f

typedef __attribute__((ext_vector_type(16))) __bf16 v16bf;
typedef __attribute__((ext_vector_type(8)))  __bf16 v8bf;
typedef __attribute__((ext_vector_type(8)))  float  v8f;
typedef __attribute__((ext_vector_type(4)))  int    i4;

#if defined(__has_builtin)
#if __has_builtin(__builtin_amdgcn_global_load_async_to_lds_b128)
#define HAVE_ASYNC_LDS 1
#endif
#endif
#ifndef HAVE_ASYNC_LDS
#define HAVE_ASYNC_LDS 0
#endif

// Builtin prototype: (V4i addrspace(1)*, V4i addrspace(3)*, Imm i32, Imm i32)
typedef __attribute__((address_space(1))) i4 gi4;
typedef __attribute__((address_space(3))) i4 li4;

__device__ __forceinline__ v16bf make_frag(uint4 lo, uint4 hi) {
  v8bf l = __builtin_bit_cast(v8bf, lo);
  v8bf h = __builtin_bit_cast(v8bf, hi);
  return __builtin_shufflevector(l, h, 0, 1, 2, 3, 4, 5, 6, 7, 8, 9, 10, 11,
                                 12, 13, 14, 15);
}

// -------------------------------------------------------------------------
// Kernel 1: per-head projections.  grid = (N/16, H*2), block = 128 (4 waves)
// Each wave computes one 16x16 tile; 4 waves cover Fp=64.
// -------------------------------------------------------------------------
__global__ void gat_proj(const float* __restrict__ X,
                         const float* __restrict__ W1,
                         const float* __restrict__ W2,
                         float* __restrict__ HW1,
                         float* __restrict__ HW2,
                         __bf16* __restrict__ HW2T) {  // [H][FP][N] bf16
  const int rowTile = blockIdx.x;        // 0..63
  const int h       = blockIdx.y >> 1;   // head
  const int which   = blockIdx.y & 1;    // 0 -> W1/HW1, 1 -> W2/HW2
  const int wave    = threadIdx.x >> 5;  // col tile 0..3
  const int lane    = threadIdx.x & 31;
  const int half    = lane >> 4;
  const int l16     = lane & 15;

  const float* W   = (which ? W2 : W1) + h * FF * FP;
  float*       out = (which ? HW2 : HW1) + h * NN * FP;

  const int row = rowTile * 16 + l16;    // A-matrix row for this lane
  const int col = wave * 16 + l16;       // B/D column for this lane

  v8f acc = {};
  #pragma unroll
  for (int kk = 0; kk < FF; kk += 32) {
    v16bf afr, bfr;
    #pragma unroll
    for (int t = 0; t < 16; ++t) {
      // ISA 7.12.2 16-bit 16x32 A layout: lanes 0-15 -> K {0..7,16..23},
      // lanes 16-31 -> K {8..15,24..31}; B mirrored with N on lanes.
      const int k = kk + 8 * half + (t & 7) + ((t >= 8) ? 16 : 0);
      afr[t] = (__bf16)X[row * FF + k];
      bfr[t] = (__bf16)W[k * FP + col];
    }
    acc = __builtin_amdgcn_wmma_f32_16x16x32_bf16(false, afr, false, bfr,
                                                  (short)0, acc, false, false);
  }
  #pragma unroll
  for (int r = 0; r < 8; ++r) {
    const int m = rowTile * 16 + r + 8 * half;  // C/D: VGPR r -> M, lane -> N
    out[m * FP + col] = acc[r];
    if (which)  // transposed bf16 copy for the aggregation GEMM B-operand
      HW2T[(h * FP + col) * NN + m] = (__bf16)acc[r];
  }
}

// -------------------------------------------------------------------------
// Kernel 2: edge-masked attention logits + sigmoid + normalization.
// grid = (N, H), block = 256.  Non-edges (A==0) give sigmoid(-1e16)==0, so
// the 64-wide lrelu dot runs only on ~5% of pairs.
// -------------------------------------------------------------------------
__global__ void gat_att(const float* __restrict__ A,
                        const float* __restrict__ HW1,
                        const float* __restrict__ HW2,
                        const float* __restrict__ bvec,
                        const float* __restrict__ a_att,
                        float* __restrict__ att_out,    // [H][N][N] final att
                        __bf16* __restrict__ att_bf,    // bf16 copy for GEMM
                        float* __restrict__ beta_out)   // [H][N]
{
  const int i   = blockIdx.x;
  const int h   = blockIdx.y;
  const int tid = threadIdx.x;

  __shared__ float ub[FP];     // HW1[i,:] + bias
  __shared__ float aw[FP];     // a_att[h,:]
  __shared__ float red[256];

  if (tid < FP) {
    ub[tid] = HW1[(h * NN + i) * FP + tid] + bvec[h * FP + tid];
    aw[tid] = a_att[h * FP + tid];
  }
  __syncthreads();

  const float* Arow = A + i * NN;
  const float* HW2h = HW2 + h * NN * FP;

  float mloc[4], attv[4];
  #pragma unroll
  for (int s = 0; s < 4; ++s) {
    const int j = tid + s * 256;
    mloc[s] = Arow[j];
    if (mloc[s] != 0.0f)                       // warm caches for edge rows
      __builtin_prefetch(HW2h + j * FP, 0, 0); // -> global_prefetch_b8
  }

  float lsum = 0.0f;
  #pragma unroll
  for (int s = 0; s < 4; ++s) {
    const int j = tid + s * 256;
    float av = 0.0f;
    if (mloc[s] != 0.0f) {
      const float4* v = (const float4*)(HW2h + j * FP);
      float logit = 0.0f;
      #pragma unroll
      for (int e4 = 0; e4 < FP / 4; ++e4) {
        const float4 vv = v[e4];
        float p0 = ub[e4 * 4 + 0] + vv.x;
        float p1 = ub[e4 * 4 + 1] + vv.y;
        float p2 = ub[e4 * 4 + 2] + vv.z;
        float p3 = ub[e4 * 4 + 3] + vv.w;
        p0 = (p0 > 0.0f) ? p0 : SLOPE * p0;
        p1 = (p1 > 0.0f) ? p1 : SLOPE * p1;
        p2 = (p2 > 0.0f) ? p2 : SLOPE * p2;
        p3 = (p3 > 0.0f) ? p3 : SLOPE * p3;
        logit = fmaf(aw[e4 * 4 + 0], p0, logit);
        logit = fmaf(aw[e4 * 4 + 1], p1, logit);
        logit = fmaf(aw[e4 * 4 + 2], p2, logit);
        logit = fmaf(aw[e4 * 4 + 3], p3, logit);
      }
      av = 1.0f / (1.0f + __expf(-logit));    // sigmoid attention
    }
    attv[s] = av;
    lsum += av;
  }

  // row sum across the block
  red[tid] = lsum;
  __syncthreads();
  #pragma unroll
  for (int off = 128; off > 0; off >>= 1) {
    if (tid < off) red[tid] += red[tid + off];
    __syncthreads();
  }
  const float beta = 1.0f / (1.0f + red[0]);  // promoter/self weight
  if (tid == 0) beta_out[h * NN + i] = beta;

  float*  arow   = att_out + (h * NN + i) * NN;
  __bf16* arowbf = att_bf + (h * NN + i) * NN;
  #pragma unroll
  for (int s = 0; s < 4; ++s) {
    const float a = attv[s] * beta;           // att / (1 + att_sum)
    arow[tid + s * 256]   = a;
    arowbf[tid + s * 256] = (__bf16)a;
  }
}

// -------------------------------------------------------------------------
// Kernel 3: out = relu(beta*HW1 + att @ HW2), head-concat layout [N][H*Fp].
// grid = (N/16, H), block = 128 (4 waves); K = 1024 -> 32 WMMA per wave.
// B tile (HW2^T slice, 64 cols x 32 K bf16 = 4KB) is shared by all 4 waves:
// staged into LDS with async global->LDS copies, double-buffered.
// -------------------------------------------------------------------------
__global__ void gat_agg(const __bf16* __restrict__ attbf,  // [H][N][N]
                        const __bf16* __restrict__ HW2T,   // [H][FP][N]
                        const float* __restrict__ HW1,
                        const float* __restrict__ beta,
                        float* __restrict__ out)           // [N][H*FP]
{
  const int rowTile = blockIdx.x;
  const int h       = blockIdx.y;
  const int wave    = threadIdx.x >> 5;
  const int lane    = threadIdx.x & 31;
  const int half    = lane >> 4;
  const int l16     = lane & 15;
  const int tid     = threadIdx.x;

  __shared__ __bf16 bt[2][FP * 32];   // [buf][col][k] bf16, 4KB per buffer

  const __bf16* attH  = attbf + h * NN * NN;
  const __bf16* hw2tH = HW2T + h * FP * NN;
  const int row = rowTile * 16 + l16;
  const int col = wave * 16 + l16;

  // Stage one 64x32 bf16 B chunk into LDS: 256 x 16B, 2 segments per thread.
  auto stage = [&](int buf, int kk) {
    #pragma unroll
    for (int s = 0; s < 2; ++s) {
      const int seg = tid + s * 128;        // 0..255
      const int c   = seg >> 2;             // column 0..63
      const int ko  = (seg & 3) * 8;        // k offset in elements
      const __bf16* g = hw2tH + c * NN + kk + ko;
      __bf16*       l = &bt[buf][c * 32 + ko];
#if HAVE_ASYNC_LDS
      __builtin_amdgcn_global_load_async_to_lds_b128((gi4*)g, (li4*)l, 0, 0);
#else
      *(uint4*)l = *(const uint4*)g;
#endif
    }
  };

  stage(0, 0);

  v8f acc = {};
  for (int c = 0; c < NN / 32; ++c) {
    const int kk  = c * 32;
    const int buf = c & 1;
#if HAVE_ASYNC_LDS
    asm volatile("s_wait_asynccnt 0x0" ::: "memory");
#endif
    __syncthreads();
    if (c + 1 < NN / 32) stage(buf ^ 1, kk + 32);

    // A fragment: two contiguous 16B loads per lane from bf16 att.
    const __bf16* ar = attH + row * NN + kk + 8 * half;
    const uint4 alo = *(const uint4*)(ar);
    const uint4 ahi = *(const uint4*)(ar + 16);
    // B fragment: two contiguous 16B loads per lane from LDS (ds_load_b128).
    const __bf16* br = &bt[buf][col * 32 + 8 * half];
    const uint4 blo = *(const uint4*)(br);
    const uint4 bhi = *(const uint4*)(br + 16);

    const v16bf afr = make_frag(alo, ahi);
    const v16bf bfr = make_frag(blo, bhi);
    acc = __builtin_amdgcn_wmma_f32_16x16x32_bf16(false, afr, false, bfr,
                                                  (short)0, acc, false, false);
    __syncthreads();  // all waves done with buf before it is re-staged
  }

  #pragma unroll
  for (int r = 0; r < 8; ++r) {
    const int m = rowTile * 16 + r + 8 * half;
    const float val =
        beta[h * NN + m] * HW1[(h * NN + m) * FP + col] + acc[r];
    out[m * (HH * FP) + h * FP + col] = fmaxf(val, 0.0f);  // relu + concat
  }
}

// -------------------------------------------------------------------------
extern "C" void kernel_launch(void* const* d_in, const int* in_sizes, int n_in,
                              void* d_out, int out_size, void* d_ws,
                              size_t ws_size, hipStream_t stream) {
  const float* X    = (const float*)d_in[0];  // [1,N,F]
  const float* A    = (const float*)d_in[1];  // [1,N,N]
  const float* W1   = (const float*)d_in[2];  // [H,F,Fp]
  const float* W2   = (const float*)d_in[3];  // [H,F,Fp]
  const float* bv   = (const float*)d_in[4];  // [H,Fp]
  const float* aatt = (const float*)d_in[5];  // [H,Fp]

  float* out     = (float*)d_out;             // [N][H*Fp] = 131072 floats
  float* att_out = out + NN * HH * FP;        // [H][N][N] = 2097152 floats

  float*  ws    = (float*)d_ws;
  float*  HW1   = ws;                         // H*N*Fp f32
  float*  HW2   = ws + HH * NN * FP;          // H*N*Fp f32
  float*  beta  = ws + 2 * HH * NN * FP;      // H*N f32
  __bf16* attbf = (__bf16*)(beta + HH * NN);  // H*N*N bf16 (16B aligned)
  __bf16* hw2t  = attbf + HH * NN * NN;       // H*Fp*N bf16

  gat_proj<<<dim3(NN / 16, HH * 2), dim3(128), 0, stream>>>(X, W1, W2, HW1,
                                                            HW2, hw2t);
  gat_att<<<dim3(NN, HH), dim3(256), 0, stream>>>(A, HW1, HW2, bv, aatt,
                                                  att_out, attbf, beta);
  gat_agg<<<dim3(NN / 16, HH), dim3(128), 0, stream>>>(attbf, hw2t, HW1, beta,
                                                       out);
}
